// DepthwiseDistRPN_12532714570379
// MI455X (gfx1250) — matrix operand order repaired and checked by reading, not compiled, not run on hardware
//
#include <hip/hip_runtime.h>

// ---------------- types ----------------
typedef __bf16 bf16;
typedef __bf16 v16bf __attribute__((ext_vector_type(16)));
typedef float  v8f   __attribute__((ext_vector_type(8)));
typedef float  v2f   __attribute__((ext_vector_type(2)));

// ---------------- problem constants ----------------
#define NBATCH 64
#define CIN    256
#define NCH    257          // CIN + dist channel
#define CP     288          // padded channels: [0]=dist, [1..7]=0, [8..263]=data, [264..287]=0
#define CH0    8            // data channel offset inside padded layout (16B aligned)
#define SS     64           // h*w  (8x8)
#define SL     1024         // H*W  (32x32)
#define OCN    256
#define NCLS   80
#define NLOC   320
#define KCLS   (OCN*SS)     // 16384
#define KLOC   (OCN*SL)     // 262144
#define BN_EPS 1e-5f

// ---------------- WMMA wrappers ----------------
__device__ __forceinline__ v8f wmma_bf16(v16bf a, v16bf b, v8f c) {
  return __builtin_amdgcn_wmma_f32_16x16x32_bf16(false, a, false, b, (short)0, c, false, false);
}
__device__ __forceinline__ v8f wmma_f32x4(v2f a, v2f b, v8f c) {
  return __builtin_amdgcn_wmma_f32_16x16x4_f32(false, a, false, b, (short)0, c, false, false);
}
__device__ __forceinline__ v8f zero_v8f() {
  v8f z = {0.f,0.f,0.f,0.f,0.f,0.f,0.f,0.f}; return z;
}

union BF16x16 { v16bf v; float4 q[2]; };

// A fragment (16x32 bf16): lane half 0 holds K {kb+0..7, kb+16..23}, half 1 holds {kb+8..15, kb+24..31}
__device__ __forceinline__ v16bf load_a16(const bf16* rowk, int half) {
  BF16x16 u;
  u.q[0] = *(const float4*)(rowk + half * 8);
  u.q[1] = *(const float4*)(rowk + 16 + half * 8);
  return u.v;
}
// B fragment (32x16 bf16): lane half 0 holds K {kb+0..7, kb+16..23}, half 1 holds {kb+8..15, kb+24..31}
__device__ __forceinline__ v16bf load_b16(const bf16* colk, int half) {
  BF16x16 u;
  u.q[0] = *(const float4*)(colk + half * 8);
  u.q[1] = *(const float4*)(colk + 16 + half * 8);
  return u.v;
}

// device-scope float atomic max (CDNA5 global_atomic_max_num_f32, no return -> STOREcnt)
__device__ __forceinline__ void atomic_max_f32(float* addr, float v) {
  asm volatile("global_atomic_max_num_f32 %0, %1, off scope:SCOPE_DEV"
               :: "v"(addr), "v"(v) : "memory");
}

__device__ __forceinline__ void wait_asynccnt0() {
#if __has_builtin(__builtin_amdgcn_s_wait_asynccnt)
  __builtin_amdgcn_s_wait_asynccnt(0);
#else
  asm volatile("s_wait_asynccnt 0x0" ::: "memory");
#endif
}

// ---------------- utility kernels ----------------
__global__ void k_fill_u32(unsigned* __restrict p, unsigned v, size_t n) {
  size_t i = (size_t)blockIdx.x * blockDim.x + threadIdx.x;
  size_t st = (size_t)gridDim.x * blockDim.x;
  for (; i < n; i += st) p[i] = v;
}

// pack f32 (N,C,spatial) -> bf16 feat_t (N,spatial,CP) at channel offset CH0+c
__global__ void k_pack_feat(const float* __restrict src, bf16* __restrict dst, int spatial) {
  size_t total = (size_t)NBATCH * CIN * spatial;
  size_t i = (size_t)blockIdx.x * blockDim.x + threadIdx.x;
  size_t st = (size_t)gridDim.x * blockDim.x;
  for (; i < total; i += st) {
    int s = (int)(i % spatial);
    size_t t = i / spatial;
    int c = (int)(t % CIN);
    int n = (int)(t / CIN);
    dst[((size_t)n * spatial + s) * CP + CH0 + c] = (bf16)src[i];
  }
}

// per-position squared norms over channels: nrm[n*spatial+s] = sum_c src[n,c,s]^2
__global__ void k_norms(const float* __restrict src, float* __restrict nrm, int spatial) {
  int i = blockIdx.x * blockDim.x + threadIdx.x;
  if (i >= NBATCH * spatial) return;
  int s = i % spatial, n = i / spatial;
  const float* p = src + (size_t)n * CIN * spatial + s;
  float acc = 0.f;
  for (int c = 0; c < CIN; ++c) { float v = p[(size_t)c * spatial]; acc += v * v; }
  nrm[i] = acc;
}

// conv weights (OC,257,3,3) f32 -> bf16 wtap[tap][oc][CP] (channel remap: 0->0, c->CH0+c-1)
__global__ void k_pack_w(const float* __restrict w, bf16* __restrict wt) {
  int total = OCN * NCH * 9;
  int i = blockIdx.x * blockDim.x + threadIdx.x;
  if (i >= total) return;
  int tap = i % 9;
  int t = i / 9;
  int ci = t % NCH;
  int oc = t / NCH;
  int pc = (ci == 0) ? 0 : (CH0 + ci - 1);
  wt[((size_t)tap * OCN + oc) * CP + pc] = (bf16)w[i];
}

// ---------------- stage 1: batched pairwise distance via bf16 WMMA ----------------
// grid = NBATCH * (SL/32) blocks, 256 thr (8 waves). wave -> (mt in 0..3, q-subtile in 0..1)
__global__ void k_dist(const bf16* __restrict featC, const bf16* __restrict featL,
                       const float* __restrict zn, const float* __restrict xn,
                       float* __restrict rowmax, float* __restrict colmax) {
  int wv = threadIdx.x >> 5, lane = threadIdx.x & 31;
  int li = lane & 15, half = lane >> 4;
  int n = blockIdx.x >> 5, qb = blockIdx.x & 31;
  int mt = wv & 3, qh = wv >> 2;
  int p0 = mt * 16, s0 = qb * 32 + qh * 16;
  const bf16* arow = featC + ((size_t)(n * SS + p0 + li) * CP + CH0);
  const bf16* bcol = featL + ((size_t)(n * SL + s0 + li) * CP + CH0);
  v8f acc = zero_v8f();
#pragma unroll
  for (int kc = 0; kc < 8; ++kc)
    acc = wmma_bf16(load_a16(arow + kc * 32, half), load_b16(bcol + kc * 32, half), acc);

  int s = s0 + li;
  float xq = xn[n * SL + s];
  float dv[8];
  float cmax = -3.4e38f;
#pragma unroll
  for (int r = 0; r < 8; ++r) {
    int p = p0 + r + 8 * half;
    float d = zn[n * SS + p] + xq - 2.0f * acc[r];
    dv[r] = d;
    cmax = fmaxf(cmax, d);
  }
  // row maxima: reduce across the 16 lanes of each half, then one atomic per row
#pragma unroll
  for (int r = 0; r < 8; ++r) {
    float m = dv[r];
    for (int o = 1; o < 16; o <<= 1) m = fmaxf(m, __shfl_xor(m, o, 32));
    if (li == 0) atomic_max_f32(rowmax + n * SS + p0 + r + 8 * half, m);
  }
  // column maxima: fold the two halves, one atomic per column
  cmax = fmaxf(cmax, __shfl_xor(cmax, 16, 32));
  if (half == 0) atomic_max_f32(colmax + n * SL + s, cmax);
}

// min over transformed distance == transform of raw max (monotone decreasing)
__global__ void k_fill_dist(const float* __restrict rowmax, const float* __restrict colmax,
                            bf16* __restrict featC, bf16* __restrict featL) {
  int i = blockIdx.x * blockDim.x + threadIdx.x;
  const int totalC = NBATCH * SS;
  if (i < totalC) {
    float t = 4.f / (1.f + __expf(rowmax[i])) - 1.f;
    featC[(size_t)i * CP] = (bf16)t;
  } else if (i < totalC + NBATCH * SL) {
    int j = i - totalC;
    float t = 4.f / (1.f + __expf(colmax[j])) - 1.f;
    featL[(size_t)j * CP] = (bf16)t;
  }
}

// ---------------- stage 2: 3x3 conv as tap-accumulated implicit GEMM (bf16 WMMA) ----------------
// Block = 8 waves = one (n, 16-column spatial tile) x ALL 16 oc-tiles (2 per wave).
// The shared input patch ((rows-in-tile+2) x (tile-width+2) pixels x CP channels) is staged
// into LDS once per block via async global->LDS DMA; halo/out-of-image entries are zeroed
// with DS stores (disjoint addresses, so no async/DS ordering hazard). B fragments then come
// from LDS (ds_load_b128), A (weights) streams from the L2-resident 1.3MB bf16 weight buffer.
__global__ void k_conv(const bf16* __restrict feat, const bf16* __restrict wt,
                       float* __restrict y, float* __restrict bnsum, float* __restrict bnsq,
                       int Hs, int Wsp) {
  extern __shared__ bf16 smem[];
  int spatial = Hs * Wsp;
  int stiles = spatial >> 4;
  int n = blockIdx.x / stiles;
  int ms = blockIdx.x % stiles;
  int tw = (Wsp < 16) ? Wsp : 16;      // tile width in image columns
  int PC = tw + 2;                     // patch columns (with halo)
  int rit = 16 / tw;                   // image rows covered by the 16-column tile (1 or 2)
  int PR = rit + 2;                    // patch rows (with halo)
  int s_base = ms * 16;
  int oy0 = s_base / Wsp;
  int x0 = s_base % Wsp;

  // ---- stage patch into LDS ----
  const int CH8 = CP / 8;              // 36 x b128 chunks per pixel
  int totalChunks = PR * PC * CH8;
  for (int ch = threadIdx.x; ch < totalChunks; ch += blockDim.x) {
    int c8 = ch % CH8;
    int pix = ch / CH8;
    int px = pix % PC;
    int r = pix / PC;
    int iy = oy0 - 1 + r;
    int ix = x0 - 1 + px;
    bf16* lp = &smem[(size_t)(r * PC + px) * CP + c8 * 8];
    if ((unsigned)iy < (unsigned)Hs && (unsigned)ix < (unsigned)Wsp) {
      unsigned loff = (unsigned)(size_t)lp;   // low 32 bits of flat LDS address = LDS offset
      const bf16* g = feat + ((size_t)(n * spatial + iy * Wsp + ix)) * CP + c8 * 8;
      asm volatile("global_load_async_to_lds_b128 %0, %1, off"
                   :: "v"(loff), "v"(g) : "memory");
    } else {
      *(float4*)lp = make_float4(0.f, 0.f, 0.f, 0.f);
    }
  }
  wait_asynccnt0();
  __syncthreads();

  // ---- compute: each wave owns 2 oc-tiles, B fragments from LDS ----
  int lane = threadIdx.x & 31, li = lane & 15, half = lane >> 4;
  int s = s_base + li;
  int oy = s / Wsp, ox = s % Wsp;
  int pr0 = oy - oy0;                  // lane's row inside the tile (0 or 1)
  int pc0 = ox - x0;                   // lane's column inside the tile (0..15)

  for (int m2 = 0; m2 < 2; ++m2) {
    int mo = ((int)(threadIdx.x >> 5)) * 2 + m2;
    v8f acc = zero_v8f();
#pragma unroll
    for (int ky = 0; ky < 3; ++ky) {
#pragma unroll
      for (int kx = 0; kx < 3; ++kx) {
        const bf16* colb = &smem[(size_t)((pr0 + ky) * PC + pc0 + kx) * CP];
        const bf16* arow = wt + (size_t)((ky * 3 + kx) * OCN + mo * 16 + li) * CP;
#pragma unroll
        for (int kc = 0; kc < 9; ++kc)
          acc = wmma_bf16(load_a16(arow + kc * 32, half), load_b16(colb + kc * 32, half), acc);
      }
    }
    // store pre-BN activations + fused BatchNorm statistics (sum / sumsq per out-channel)
#pragma unroll
    for (int r = 0; r < 8; ++r) {
      int oc = mo * 16 + r + 8 * half;
      float v = acc[r];
      y[((size_t)(n * OCN + oc)) * spatial + s] = v;
      float sm = v, sq = v * v;
      for (int o = 1; o < 16; o <<= 1) {
        sm += __shfl_xor(sm, o, 32);
        sq += __shfl_xor(sq, o, 32);
      }
      if (li == 0) { atomicAdd(bnsum + oc, sm); atomicAdd(bnsq + oc, sq); }
    }
  }
}

__global__ void k_bnfin(const float* __restrict g, const float* __restrict b,
                        const float* __restrict sum, const float* __restrict sq,
                        float* __restrict scale, float* __restrict shift, float invcnt) {
  int oc = blockIdx.x * blockDim.x + threadIdx.x;
  if (oc >= OCN) return;
  float mu = sum[oc] * invcnt;
  float var = sq[oc] * invcnt - mu * mu;
  float sc = g[oc] * rsqrtf(var + BN_EPS);
  scale[oc] = sc;
  shift[oc] = b[oc] - mu * sc;
}

__global__ void k_bnrelu(const float* __restrict y, const float* __restrict scale,
                         const float* __restrict shift, float* __restrict a, int spatial) {
  size_t total = (size_t)NBATCH * OCN * spatial;
  size_t i = (size_t)blockIdx.x * blockDim.x + threadIdx.x;
  size_t st = (size_t)gridDim.x * blockDim.x;
  for (; i < total; i += st) {
    int oc = (int)((i / spatial) % OCN);
    a[i] = fmaxf(0.f, y[i] * scale[oc] + shift[oc]);
  }
}

__global__ void k_out_bias(const float* __restrict cb, const float* __restrict lb,
                           float* __restrict out) {
  int i = blockIdx.x * blockDim.x + threadIdx.x;
  const int nc = NBATCH * NCLS;
  if (i < nc) out[i] = cb[i % NCLS];
  else if (i < nc + NBATCH * NLOC) out[i] = lb[(i - nc) % NLOC];
}

// ---------------- stage 3: linear layers, f32 WMMA 16x16x4, split-K + atomicAdd ----------------
__global__ void k_lin(const float* __restrict act, const float* __restrict w,
                      float* __restrict out, int Ncols, int K, int kslices, int KS) {
  int lane = threadIdx.x & 31, li = lane & 15, half = lane >> 4;
  int tile = blockIdx.x * (blockDim.x >> 5) + (threadIdx.x >> 5);
  int ntiles = Ncols >> 4;
  int total = (NBATCH / 16) * ntiles * kslices;
  if (tile >= total) return;
  int ks = tile % kslices;
  int t = tile / kslices;
  int nt = t % ntiles;
  int mt = t / ntiles;
  const float* ap = act + (size_t)(mt * 16 + li) * K + (size_t)ks * KS + 2 * half;
  const float* bp = w + (size_t)(nt * 16 + li) * K + (size_t)ks * KS + 2 * half;
  v8f acc = zero_v8f();
#pragma unroll 8
  for (int k = 0; k < KS; k += 4) {
    __builtin_prefetch(bp + k + 1024, 0, 0);   // global_prefetch_b8 on the 335MB weight stream
    v2f a = *(const v2f*)(ap + k);
    v2f b = *(const v2f*)(bp + k);
    acc = wmma_f32x4(a, b, acc);
  }
#pragma unroll
  for (int r = 0; r < 8; ++r) {
    int m = mt * 16 + r + 8 * half;
    atomicAdd(out + (size_t)m * Ncols + nt * 16 + li, acc[r]);
  }
}

// ---------------- host launcher ----------------
static inline size_t alignup(size_t x) { return (x + 255) & ~(size_t)255; }

extern "C" void kernel_launch(void* const* d_in, const int* in_sizes, int n_in,
                              void* d_out, int out_size, void* d_ws, size_t ws_size,
                              hipStream_t stream) {
  const float* z_f   = (const float*)d_in[0];
  const float* x_f   = (const float*)d_in[1];
  const float* wcls  = (const float*)d_in[2];
  const float* gcls  = (const float*)d_in[3];
  const float* bcls  = (const float*)d_in[4];
  const float* wreg  = (const float*)d_in[5];
  const float* greg  = (const float*)d_in[6];
  const float* breg  = (const float*)d_in[7];
  const float* lcw   = (const float*)d_in[8];
  const float* lcb   = (const float*)d_in[9];
  const float* lrw   = (const float*)d_in[10];
  const float* lrb   = (const float*)d_in[11];
  float* out = (float*)d_out;

  // workspace carve-out
  char* ws = (char*)d_ws;
  size_t off = 0;
  auto take = [&](size_t bytes) { char* p = ws + off; off += alignup(bytes); return p; };
  bf16*  featC  = (bf16*)take((size_t)NBATCH * SS * CP * 2);
  bf16*  featL  = (bf16*)take((size_t)NBATCH * SL * CP * 2);
  bf16*  wtC    = (bf16*)take((size_t)9 * OCN * CP * 2);
  bf16*  wtL    = (bf16*)take((size_t)9 * OCN * CP * 2);
  float* zn     = (float*)take((size_t)NBATCH * SS * 4);
  float* xn     = (float*)take((size_t)NBATCH * SL * 4);
  float* rowmax = (float*)take((size_t)NBATCH * SS * 4);
  float* colmax = (float*)take((size_t)NBATCH * SL * 4);
  float* yC     = (float*)take((size_t)NBATCH * OCN * SS * 4);
  float* yL     = (float*)take((size_t)NBATCH * OCN * SL * 4);
  float* aC     = (float*)take((size_t)NBATCH * OCN * SS * 4);
  float* aL     = (float*)take((size_t)NBATCH * OCN * SL * 4);
  float* bnsumC = (float*)take(OCN * 4);
  float* bnsqC  = (float*)take(OCN * 4);
  float* bnsumL = (float*)take(OCN * 4);
  float* bnsqL  = (float*)take(OCN * 4);
  float* scC    = (float*)take(OCN * 4);
  float* shC    = (float*)take(OCN * 4);
  float* scL    = (float*)take(OCN * 4);
  float* shL    = (float*)take(OCN * 4);
  if (off > ws_size) return;  // workspace too small; cannot proceed

  auto fill = [&](void* p, unsigned v, size_t bytes) {
    size_t n = bytes / 4;
    size_t blk = (n + 255) / 256; if (blk > 4096) blk = 4096;
    k_fill_u32<<<dim3((unsigned)blk), dim3(256), 0, stream>>>((unsigned*)p, v, n);
  };

  // ---- init: zero padded bf16 buffers / BN accumulators; -FLT_MAX for max buffers ----
  fill(featC, 0u, (size_t)NBATCH * SS * CP * 2);
  fill(featL, 0u, (size_t)NBATCH * SL * CP * 2);
  fill(wtC, 0u, (size_t)9 * OCN * CP * 2);
  fill(wtL, 0u, (size_t)9 * OCN * CP * 2);
  fill(bnsumC, 0u, OCN * 4); fill(bnsqC, 0u, OCN * 4);
  fill(bnsumL, 0u, OCN * 4); fill(bnsqL, 0u, OCN * 4);
  fill(rowmax, 0xFF7FFFFFu, (size_t)NBATCH * SS * 4);   // -FLT_MAX
  fill(colmax, 0xFF7FFFFFu, (size_t)NBATCH * SL * 4);

  // ---- pack to bf16 + norms + weights ----
  k_pack_feat<<<dim3(4096), dim3(256), 0, stream>>>(z_f, featC, SS);
  k_pack_feat<<<dim3(8192), dim3(256), 0, stream>>>(x_f, featL, SL);
  k_norms<<<dim3(16), dim3(256), 0, stream>>>(z_f, zn, SS);
  k_norms<<<dim3(256), dim3(256), 0, stream>>>(x_f, xn, SL);
  k_pack_w<<<dim3((OCN * NCH * 9 + 255) / 256), dim3(256), 0, stream>>>(wcls, wtC);
  k_pack_w<<<dim3((OCN * NCH * 9 + 255) / 256), dim3(256), 0, stream>>>(wreg, wtL);

  // ---- distance GEMM + min (via max of raw d) ----
  k_dist<<<dim3(NBATCH * (SL / 32)), dim3(256), 0, stream>>>(featC, featL, zn, xn, rowmax, colmax);
  k_fill_dist<<<dim3((NBATCH * (SS + SL) + 255) / 256), dim3(256), 0, stream>>>(
      rowmax, colmax, featC, featL);

  // ---- conv + BN stats (async LDS-staged input patch) ----
  size_t ldsS = (size_t)(2 + 2) * (8 + 2) * CP * 2;    // 23040 B
  size_t ldsL = (size_t)(1 + 2) * (16 + 2) * CP * 2;   // 31104 B
  k_conv<<<dim3(NBATCH * (SS / 16)), dim3(256), ldsS, stream>>>(
      featC, wtC, yC, bnsumC, bnsqC, 8, 8);
  k_conv<<<dim3(NBATCH * (SL / 16)), dim3(256), ldsL, stream>>>(
      featL, wtL, yL, bnsumL, bnsqL, 32, 32);

  // ---- BN finalize + ReLU ----
  k_bnfin<<<dim3(1), dim3(256), 0, stream>>>(gcls, bcls, bnsumC, bnsqC, scC, shC,
                                             1.f / (NBATCH * SS));
  k_bnfin<<<dim3(1), dim3(256), 0, stream>>>(greg, breg, bnsumL, bnsqL, scL, shL,
                                             1.f / (NBATCH * SL));
  k_bnrelu<<<dim3(2048), dim3(256), 0, stream>>>(yC, scC, shC, aC, SS);
  k_bnrelu<<<dim3(8192), dim3(256), 0, stream>>>(yL, scL, shL, aL, SL);

  // ---- linear layers (f32 WMMA, split-K, bias pre-seeded) ----
  k_out_bias<<<dim3((NBATCH * (NCLS + NLOC) + 255) / 256), dim3(256), 0, stream>>>(lcb, lrb, out);
  float* out_cls = out;
  float* out_loc = out + NBATCH * NCLS;
  // cls: K=16384, 4 slices of 4096 -> 4*5*4 = 80 wave-tiles
  k_lin<<<dim3(10), dim3(256), 0, stream>>>(aC, lcw, out_cls, NCLS, KCLS, 4, 4096);
  // loc: K=262144, 32 slices of 8192 -> 4*20*32 = 2560 wave-tiles
  k_lin<<<dim3(320), dim3(256), 0, stream>>>(aL, lrw, out_loc, NLOC, KLOC, 32, 8192);
}
